// TimingGNN_21354577395815
// MI455X (gfx1250) — matrix operand order repaired and checked by reading, not compiled, not run on hardware
//
#include <hip/hip_runtime.h>
#include <math.h>

// ---------------------------------------------------------------------------
// CDNA5 (gfx1250) WMMA helpers: V_WMMA_F32_16X16X32_BF16, wave32.
// ---------------------------------------------------------------------------
typedef __attribute__((ext_vector_type(16))) __bf16 v16bf;
typedef __attribute__((ext_vector_type(4)))  __bf16 v4bf;
typedef __attribute__((ext_vector_type(8)))  float  v8f;

// A fragment: 16x32 bf16 tile from LDS activation buffer X (row stride ld).
// ISA layout: lane L holds row m=L&15; lanes 0-15 K={0..7,16..23}, lanes 16-31
// K={8..15,24..31}. Two contiguous 8-element groups -> 2x ds_load_b128.
__device__ __forceinline__ v16bf lds_load_a(const __bf16* X, int row0, int ld,
                                            int k0, int lane) {
  const int m  = lane & 15;
  const int h8 = (lane >> 4) << 3;
  const __bf16* p = X + (row0 + m) * ld + k0 + h8;
  v16bf a;
#pragma unroll
  for (int e = 0; e < 8; ++e) a[e] = p[e];         // K = h8 + e
#pragma unroll
  for (int e = 8; e < 16; ++e) a[e] = p[e + 8];    // K = h8 + 16 + (e-8)
  return a;
}

__device__ __forceinline__ v8f wmma_bf16(v16bf a, v16bf b, v8f c) {
  // (neg_a, A, neg_b, B, c_mod, C, reuse_a, reuse_b)
  return __builtin_amdgcn_wmma_f32_16x16x32_bf16(false, a, false, b,
                                                 (short)0, c, false, false);
}

__device__ __forceinline__ float sig_(float x) { return 1.0f / (1.0f + __expf(-x)); }
// leaky_relu(x, 0.2) == max(x, 0.2x) since slope < 1.
__device__ __forceinline__ float leaky_(float x) { return fmaxf(x, 0.2f * x); }

// Stage weight matrix W (K x Nsrc, f32 row-major) into fragment-major bf16 LDS:
// fragment (ks, t) holds the 32x16 B tile for k in [32ks,32ks+32), n in
// [16t,16t+16).  Per-lane 16 elements stored contiguously so a B-fragment load
// is a single v16bf (2x ds_load_b128).  B ISA layout: lane = 16*((kr>>3)&1) +
// (n&15), element e -> K = 32ks + 8*half + (e&7) + 16*(e>>3).
__device__ __forceinline__ void stage_w_frag(v16bf* sW, const float* W, int K,
                                             int Nsrc, int Ndst, int tid,
                                             int nthreads) {
  __bf16* w = (__bf16*)sW;
  const int NT = Ndst >> 4;
  for (int i = tid; i < K * Ndst; i += nthreads) {
    int k = i / Ndst, n = i - k * Ndst;
    float v = (n < Nsrc) ? W[k * Nsrc + n] : 0.0f;
    int ks   = k >> 5, kr = k & 31;
    int lane = ((kr & 8) << 1) | (n & 15);
    int e    = (kr & 7) | ((kr & 16) >> 1);
    int t    = n >> 4;
    w[((((ks * NT + t) << 5) + lane) << 4) + e] = (__bf16)v;
  }
}

// ---------------------------------------------------------------------------
// Utility kernels
// ---------------------------------------------------------------------------
__global__ void gnn_zero(float* __restrict__ p, size_t n) {
  size_t i = (size_t)blockIdx.x * blockDim.x + threadIdx.x;
  if (i < n) p[i] = 0.0f;
}

__global__ void gnn_degree(const int* __restrict__ dst, float* __restrict__ deg, int E) {
  int e = blockIdx.x * blockDim.x + threadIdx.x;
  if (e < E) atomicAdd(&deg[dst[e]], 1.0f);
}

// Bias + leaky-relu + bf16 store of one C tile (8 rows per lane).
__device__ __forceinline__ void store_act(__bf16* X, int ld, int mb, int col,
                                          v8f c, float bias) {
#pragma unroll
  for (int r = 0; r < 8; ++r) {
    float v = leaky_(c[r] + bias);
    X[(mb + r) * ld + col] = (__bf16)v;
  }
}

// ---------------------------------------------------------------------------
// Edge MLP: x = leaky(leaky([inA[src]|inB[src]] @ W0 + b0) @ W1 + b1) @ W2 + b2
// gate = sigmoid(x[:,0]); atomic-accumulate x[:,1:33]*gate -> acc1[dst],
// x[:,33:65]*gate -> acc2[dst].  128 edges/block, 8 waves.
// ---------------------------------------------------------------------------
#define ETM 128

__global__ __launch_bounds__(256) void gnn_edge_mlp(
    const int* __restrict__ srcI, const int* __restrict__ dstI,
    const float* __restrict__ inA, const float* __restrict__ inB,
    const float* __restrict__ W0, const float* __restrict__ B0,
    const float* __restrict__ W1, const float* __restrict__ B1,
    const float* __restrict__ W2, const float* __restrict__ B2,
    float* __restrict__ acc1, float* __restrict__ acc2, int E) {
  __shared__ __align__(16) __bf16 sX0[ETM * 64];
  __shared__ __align__(16) __bf16 sX1[ETM * 64];
  __shared__ v16bf sW0f[2 * 4 * 32];   // K=64 (2 ksteps), N=64 (4 tiles)
  __shared__ v16bf sW1f[2 * 4 * 32];
  __shared__ v16bf sW2f[2 * 5 * 32];   // N=65 padded to 80 (5 tiles)
  __shared__ float sB0[64], sB1[64], sB2[80];
  __shared__ int   sSrc[ETM], sDst[ETM];

  const int  tid  = threadIdx.x;
  const int  lane = tid & 31;
  const int  wave = tid >> 5;
  const long eBase = (long)blockIdx.x * ETM;
  const bool fullBlock = (eBase + ETM) <= (long)E;

  stage_w_frag(sW0f, W0, 64, 64, 64, tid, 256);
  stage_w_frag(sW1f, W1, 64, 64, 64, tid, 256);
  stage_w_frag(sW2f, W2, 64, 65, 80, tid, 256);
  if (tid < 64) { sB0[tid] = B0[tid]; sB1[tid] = B1[tid]; }
  if (tid < 80) sB2[tid] = (tid < 65) ? B2[tid] : 0.0f;
  if (tid < ETM) {
    long e = eBase + tid;
    bool v = e < (long)E;
    sSrc[tid] = v ? srcI[e] : 0;
    sDst[tid] = v ? dstI[e] : 0;
  }
  __syncthreads();

  // Gather edge inputs as float4, store packed bf16 (b64 per slot).
  // Row layout: cols 0..31 = inA[src], 32..63 = inB[src].
  for (int i = tid; i < ETM * 16; i += 256) {
    int  row = i >> 4, slot = i & 15;
    long s = sSrc[row];
    const float4* p = (slot < 8) ? (const float4*)(inA + s * 32) + slot
                                 : (const float4*)(inB + s * 32) + (slot - 8);
    float4 v = *p;
    v4bf b;
    b[0] = (__bf16)v.x; b[1] = (__bf16)v.y; b[2] = (__bf16)v.z; b[3] = (__bf16)v.w;
    *(v4bf*)(sX0 + row * 64 + slot * 4) = b;
  }
  __syncthreads();

  const int row0 = wave * 16;
  const int n    = lane & 15;
  const int half = lane >> 4;
  const int mb   = row0 + half * 8;

  // Layer 1: X0(128x64) @ W0 -> X1, leaky relu.
  {
    v16bf a0 = lds_load_a(sX0, row0, 64, 0, lane);
    v16bf a1 = lds_load_a(sX0, row0, 64, 32, lane);
#pragma unroll
    for (int t = 0; t < 4; ++t) {
      v8f c = {};
      c = wmma_bf16(a0, sW0f[(t)     * 32 + lane], c);
      c = wmma_bf16(a1, sW0f[(4 + t) * 32 + lane], c);
      store_act(sX1, 64, mb, t * 16 + n, c, sB0[t * 16 + n]);
    }
  }
  __syncthreads();

  // Layer 2: X1 @ W1 -> X0 (reuse), leaky relu.
  {
    v16bf a0 = lds_load_a(sX1, row0, 64, 0, lane);
    v16bf a1 = lds_load_a(sX1, row0, 64, 32, lane);
#pragma unroll
    for (int t = 0; t < 4; ++t) {
      v8f c = {};
      c = wmma_bf16(a0, sW1f[(t)     * 32 + lane], c);
      c = wmma_bf16(a1, sW1f[(4 + t) * 32 + lane], c);
      store_act(sX0, 64, mb, t * 16 + n, c, sB1[t * 16 + n]);
    }
  }
  __syncthreads();

  // Layer 3: X0 @ W2 (64x80) + b2; keep all 5 tile accumulators.
  v8f c3[5];
  {
    v16bf a0 = lds_load_a(sX0, row0, 64, 0, lane);
    v16bf a1 = lds_load_a(sX0, row0, 64, 32, lane);
#pragma unroll
    for (int t = 0; t < 5; ++t) {
      v8f c = {};
      c = wmma_bf16(a0, sW2f[(t)     * 32 + lane], c);
      c = wmma_bf16(a1, sW2f[(5 + t) * 32 + lane], c);
      float bias = sB2[t * 16 + n];
#pragma unroll
      for (int r = 0; r < 8; ++r) c[r] += bias;
      c3[t] = c;
    }
  }

  // Gate: column 0 lives at lanes {0,16} of tile 0; broadcast per row-register.
  float g[8];
  {
    const int srcl = lane & 16;
#pragma unroll
    for (int r = 0; r < 8; ++r) g[r] = sig_(__shfl(c3[0][r], srcl, 32));
  }

  // Scatter gated messages with float atomics (segment-sum numerator).
  if (fullBlock) {
#pragma unroll
    for (int t = 0; t < 5; ++t) {
      int col = t * 16 + n;
#pragma unroll
      for (int r = 0; r < 8; ++r) {
        long  d = sDst[mb + r];
        float v = c3[t][r] * g[r];
        if (col >= 1 && col <= 32)
          atomicAdd(&acc1[d * 32 + (col - 1)], v);
        else if (col >= 33 && col <= 64)
          atomicAdd(&acc2[d * 32 + (col - 33)], v);
      }
    }
  } else {
#pragma unroll
    for (int t = 0; t < 5; ++t) {
      int col = t * 16 + n;
#pragma unroll
      for (int r = 0; r < 8; ++r) {
        int m = mb + r;
        if (eBase + m < (long)E) {
          long  d = sDst[m];
          float v = c3[t][r] * g[r];
          if (col >= 1 && col <= 32)
            atomicAdd(&acc1[d * 32 + (col - 1)], v);
          else if (col >= 33 && col <= 64)
            atomicAdd(&acc2[d * 32 + (col - 33)], v);
        }
      }
    }
  }
}

// ---------------------------------------------------------------------------
// Node MLP (96->64->64->96) fused with segment-mean + sigmoid combine.
// 64 nodes/block, 4 waves.
// ---------------------------------------------------------------------------
#define NTM 64

__global__ __launch_bounds__(128) void gnn_node_finalize(
    const float* __restrict__ fw, const float* __restrict__ bw,
    const float* __restrict__ bd,
    const float* __restrict__ W0, const float* __restrict__ B0,
    const float* __restrict__ W1, const float* __restrict__ B1,
    const float* __restrict__ W2, const float* __restrict__ B2,
    const float* __restrict__ accF1, const float* __restrict__ accF2,
    const float* __restrict__ accB1, const float* __restrict__ accB2,
    const float* __restrict__ deg,
    float* __restrict__ outBD, float* __restrict__ outBW,
    float* __restrict__ outFW, int N) {
  __shared__ __align__(16) __bf16 sX0[NTM * 96];
  __shared__ __align__(16) __bf16 sX1[NTM * 64];
  __shared__ v16bf sW0f[3 * 4 * 32];   // K=96 (3 ksteps), N=64 (4 tiles)
  __shared__ v16bf sW1f[2 * 4 * 32];
  __shared__ v16bf sW2f[2 * 6 * 32];   // K=64, N=96 (6 tiles)
  __shared__ float sB0[64], sB1[64], sB2[96];

  const int  tid  = threadIdx.x;
  const int  lane = tid & 31;
  const int  wave = tid >> 5;
  const long nBase = (long)blockIdx.x * NTM;

  stage_w_frag(sW0f, W0, 96, 64, 64, tid, 128);
  stage_w_frag(sW1f, W1, 64, 64, 64, tid, 128);
  stage_w_frag(sW2f, W2, 64, 96, 96, tid, 128);
  if (tid < 64) { sB0[tid] = B0[tid]; sB1[tid] = B1[tid]; }
  if (tid < 96) sB2[tid] = B2[tid];

  // Gather node features (float4 -> packed bf16): cols 0..31 fw, 32..63 bw,
  // 64..95 bd.  Pad rows clamp to N-1 (garbage computed, never stored).
  for (int i = tid; i < NTM * 24; i += 128) {
    int  row = i / 24, slot = i - row * 24;
    long node = nBase + row;
    if (node >= N) node = N - 1;
    const float4* p = (slot < 8)  ? (const float4*)(fw + node * 32) + slot
                    : (slot < 16) ? (const float4*)(bw + node * 32) + (slot - 8)
                                  : (const float4*)(bd + node * 32) + (slot - 16);
    float4 v = *p;
    v4bf b;
    b[0] = (__bf16)v.x; b[1] = (__bf16)v.y; b[2] = (__bf16)v.z; b[3] = (__bf16)v.w;
    *(v4bf*)(sX0 + row * 96 + slot * 4) = b;
  }
  __syncthreads();

  const int row0 = wave * 16;
  const int n    = lane & 15;
  const int half = lane >> 4;
  const int mb   = row0 + half * 8;

  // Layer 1: X0(64x96) @ W0(96x64) -> X1, leaky relu (K loop = 3).
  {
    v16bf a0 = lds_load_a(sX0, row0, 96, 0, lane);
    v16bf a1 = lds_load_a(sX0, row0, 96, 32, lane);
    v16bf a2 = lds_load_a(sX0, row0, 96, 64, lane);
#pragma unroll
    for (int t = 0; t < 4; ++t) {
      v8f c = {};
      c = wmma_bf16(a0, sW0f[(t)     * 32 + lane], c);
      c = wmma_bf16(a1, sW0f[(4 + t) * 32 + lane], c);
      c = wmma_bf16(a2, sW0f[(8 + t) * 32 + lane], c);
      store_act(sX1, 64, mb, t * 16 + n, c, sB0[t * 16 + n]);
    }
  }
  __syncthreads();

  // Layer 2: X1 @ W1 -> X0 (cols 0..63, stride 96), leaky relu.
  {
    v16bf a0 = lds_load_a(sX1, row0, 64, 0, lane);
    v16bf a1 = lds_load_a(sX1, row0, 64, 32, lane);
#pragma unroll
    for (int t = 0; t < 4; ++t) {
      v8f c = {};
      c = wmma_bf16(a0, sW1f[(t)     * 32 + lane], c);
      c = wmma_bf16(a1, sW1f[(4 + t) * 32 + lane], c);
      store_act(sX0, 96, mb, t * 16 + n, c, sB1[t * 16 + n]);
    }
  }
  __syncthreads();

  // Layer 3: X0(cols 0..63, ld 96) @ W2(64x96) + b2, fused finalize.
  {
    v16bf a0 = lds_load_a(sX0, row0, 96, 0, lane);
    v16bf a1 = lds_load_a(sX0, row0, 96, 32, lane);
#pragma unroll
    for (int t = 0; t < 6; ++t) {
      v8f c = {};
      c = wmma_bf16(a0, sW2f[(t)     * 32 + lane], c);
      c = wmma_bf16(a1, sW2f[(6 + t) * 32 + lane], c);
      int   col  = t * 16 + n;
      float bias = sB2[col];
      int   j    = col & 31;
      int   sel  = col >> 5;  // 0: o_bd, 1: o_bw, 2: o_fw
#pragma unroll
      for (int r = 0; r < 8; ++r) {
        long node = nBase + mb + r;
        if (node < N) {
          float dg  = deg[node];
          float inv = 1.0f / (dg < 1.0f ? 1.0f : dg);
          float add;
          if (sel == 0)      add = (accF1[node * 32 + j] + accB1[node * 32 + j]) * inv;
          else if (sel == 1) add = accB2[node * 32 + j] * inv;
          else               add = accF2[node * 32 + j] * inv;
          float o = sig_(c[r] + bias + add);
          float* op = (sel == 0) ? outBD : (sel == 1) ? outBW : outFW;
          op[node * 32 + j] = o;
        }
      }
    }
  }
}

// ---------------------------------------------------------------------------
// Launch
// ---------------------------------------------------------------------------
extern "C" void kernel_launch(void* const* d_in, const int* in_sizes, int n_in,
                              void* d_out, int out_size, void* d_ws, size_t ws_size,
                              hipStream_t stream) {
  const int*   src = (const int*)d_in[0];
  const int*   dst = (const int*)d_in[1];
  const float* bd  = (const float*)d_in[2];
  const float* bw  = (const float*)d_in[3];
  const float* fw  = (const float*)d_in[4];
  const float* Wf0 = (const float*)d_in[5];  const float* bf0 = (const float*)d_in[6];
  const float* Wf1 = (const float*)d_in[7];  const float* bf1 = (const float*)d_in[8];
  const float* Wf2 = (const float*)d_in[9];  const float* bf2 = (const float*)d_in[10];
  const float* Wb0 = (const float*)d_in[11]; const float* bb0 = (const float*)d_in[12];
  const float* Wb1 = (const float*)d_in[13]; const float* bb1 = (const float*)d_in[14];
  const float* Wb2 = (const float*)d_in[15]; const float* bb2 = (const float*)d_in[16];
  const float* Ws0 = (const float*)d_in[17]; const float* bs0 = (const float*)d_in[18];
  const float* Ws1 = (const float*)d_in[19]; const float* bs1 = (const float*)d_in[20];
  const float* Ws2 = (const float*)d_in[21]; const float* bs2 = (const float*)d_in[22];

  const int E = in_sizes[0];
  const int N = in_sizes[2] / 32;

  float* ws    = (float*)d_ws;
  float* accF1 = ws;
  float* accF2 = ws + (size_t)N * 32;
  float* accB1 = ws + (size_t)N * 64;
  float* accB2 = ws + (size_t)N * 96;
  float* deg   = ws + (size_t)N * 128;

  float* outBD = (float*)d_out;
  float* outBW = outBD + (size_t)N * 32;
  float* outFW = outBD + (size_t)N * 64;

  const size_t zc = (size_t)N * 128 + (size_t)N;
  gnn_zero<<<(unsigned)((zc + 255) / 256), 256, 0, stream>>>(ws, zc);
  gnn_degree<<<(E + 255) / 256, 256, 0, stream>>>(dst, deg, E);

  const int eBlocks = (E + ETM - 1) / ETM;
  gnn_edge_mlp<<<eBlocks, 256, 0, stream>>>(src, dst, fw, bd,
                                            Wf0, bf0, Wf1, bf1, Wf2, bf2,
                                            accF1, accF2, E);
  gnn_edge_mlp<<<eBlocks, 256, 0, stream>>>(src, dst, bw, bd,
                                            Wb0, bb0, Wb1, bb1, Wb2, bb2,
                                            accB1, accB2, E);

  const int nBlocks = (N + NTM - 1) / NTM;
  gnn_node_finalize<<<nBlocks, 128, 0, stream>>>(fw, bw, bd,
                                                 Ws0, bs0, Ws1, bs1, Ws2, bs2,
                                                 accF1, accF2, accB1, accB2, deg,
                                                 outBD, outBW, outFW, N);
}
